// GCNModel_87050397156003
// MI455X (gfx1250) — matrix operand order repaired and checked by reading, not compiled, not run on hardware
//
#include <hip/hip_runtime.h>
#include <hip/hip_bf16.h>
#include <math.h>

// GCN forward for MI455X (gfx1250, wave32).
// Dense GEMMs: v_wmma_f32_16x16x32_f16, f32 accumulate, pre-converted f16
// operands (A row-major, B transposed) so fragments load as global_load_b128.
// Sparse edge aggregation: float4 gathers + global_atomic_add_f32 scatters.

typedef __attribute__((ext_vector_type(16))) _Float16 v16h;
typedef __attribute__((ext_vector_type(8)))  _Float16 v8h;
typedef __attribute__((ext_vector_type(4)))  _Float16 v4h;
typedef __attribute__((ext_vector_type(8)))  float    v8f;

#define HID  128
#define NGRF 256
#define OUTC 10

// ---------------- degree / normalization ----------------

__global__ void k_deg_init(float* __restrict__ deg, int n) {
    int i = blockIdx.x * blockDim.x + threadIdx.x;
    if (i < n) deg[i] = 1.0f;                       // self loop contributes 1
}

__global__ void k_deg_count(float* __restrict__ deg, const int* __restrict__ col, int e) {
    int i = blockIdx.x * blockDim.x + threadIdx.x;
    if (i < e) atomicAdd(&deg[col[i]], 1.0f);
}

__global__ void k_rsqrt_inplace(float* __restrict__ deg, int n) {
    int i = blockIdx.x * blockDim.x + threadIdx.x;
    if (i < n) deg[i] = rsqrtf(deg[i]);             // deg >= 1 always (self loop)
}

// ---------------- operand conversion ----------------

// Wt[n*128+k] = (f16) W[k*128+n]   (transpose+convert; 128x128)
__global__ void k_w_transpose_f16(const float* __restrict__ W, _Float16* __restrict__ Wt) {
    int i = blockIdx.x * blockDim.x + threadIdx.x;   // i = n*128 + k
    if (i >= HID * HID) return;
    int n = i >> 7, k = i & 127;
    Wt[i] = (_Float16)W[k * HID + n];
}

// dst f16 = src f32, 4-wide
__global__ void k_f32_to_f16(const float* __restrict__ src, _Float16* __restrict__ dst, int n4) {
    int i = blockIdx.x * blockDim.x + threadIdx.x;
    if (i >= n4) return;
    float4 v = ((const float4*)src)[i];
    v4h o = { (_Float16)v.x, (_Float16)v.y, (_Float16)v.z, (_Float16)v.w };
    ((v4h*)dst)[i] = o;
}

// ---------------- WMMA GEMM: C[M,128] = A[M,128] * W[128,128] ----------------
// A: row-major f16. Bt: transposed f16 (Bt[n][k]).
// Block = 256 threads = 8 waves = 4 row-tiles x 2 col-tiles -> 64x32 output.
// grid = (cdiv(M,64), 128/32). K=128 -> 4 x v_wmma_f32_16x16x32_f16 per wave.

__global__ __launch_bounds__(256) void k_gemm_wmma(
    const _Float16* __restrict__ A, const _Float16* __restrict__ Bt,
    float* __restrict__ C, int M) {
    const int wave = threadIdx.x >> 5;
    const int lane = threadIdx.x & 31;
    const int hi   = lane >> 4;             // 0 or 1
    const int lo   = lane & 15;
    const int rowBase = blockIdx.x * 64 + (wave & 3) * 16;
    const int colBase = blockIdx.y * 32 + (wave >> 2) * 16;

    int m = rowBase + lo;                   // clamp (keeps EXEC uniform for WMMA)
    if (m > M - 1) m = M - 1;

    const _Float16* arow = A  + (size_t)m * HID + hi * 8;               // K = hi*8 + ...
    const _Float16* bcol = Bt + (size_t)(colBase + lo) * HID + hi * 8;  // K = hi*8 + ...

    v8f acc = {};
#pragma unroll
    for (int kk = 0; kk < 128; kk += 32) {
        // A fragment: lane L -> M = L&15 ; K = kk + hi*8 + {0..7, 16..23}
        v8h a0 = *(const v8h*)(arow + kk);
        v8h a1 = *(const v8h*)(arow + kk + 16);
        v16h a = __builtin_shufflevector(a0, a1, 0,1,2,3,4,5,6,7,8,9,10,11,12,13,14,15);
        // B fragment: lane L -> N = L&15 ; K = kk + hi*8 + {0..7, 16..23}
        v8h b0 = *(const v8h*)(bcol + kk);
        v8h b1 = *(const v8h*)(bcol + kk + 16);
        v16h b = __builtin_shufflevector(b0, b1, 0,1,2,3,4,5,6,7,8,9,10,11,12,13,14,15);
        acc = __builtin_amdgcn_wmma_f32_16x16x32_f16(
            /*neg_a=*/false, a, /*neg_b=*/false, b,
            /*c_mod=*/(short)0, acc, /*reuse_a=*/false, /*reuse_b=*/false);
    }
    // D layout: acc[r] = C[rowBase + hi*8 + r][colBase + lo]
    float* crow = C + (size_t)(rowBase + hi * 8) * HID + colBase + lo;
    if (rowBase + 16 <= M) {                // full tile: straight-line stores
#pragma unroll
        for (int r = 0; r < 8; ++r) crow[(size_t)r * HID] = acc[r];
    } else {                                // ragged tail tile only
#pragma unroll
        for (int r = 0; r < 8; ++r)
            if (rowBase + hi * 8 + r < M) crow[(size_t)r * HID] = acc[r];
    }
}

// ---------------- aggregation (self loop + edges), bias+relu ----------------

// out[i][:] = xw[i][:] * dinv[i]^2   (float4 per thread, 32 threads per node)
__global__ void k_self_init(const float* __restrict__ xw, const float* __restrict__ dinv,
                            float* __restrict__ outp, int n) {
    int idx = blockIdx.x * blockDim.x + threadIdx.x;
    if (idx >= n * 32) return;
    int node = idx >> 5;
    float d = dinv[node];
    float s = d * d;
    float4 v = ((const float4*)xw)[idx];
    v.x *= s; v.y *= s; v.z *= s; v.w *= s;
    ((float4*)outp)[idx] = v;
}

// one wave32 per edge: gather xw[row] (float4/lane), scale, atomic-scatter to col
__global__ void k_edge_agg(const float* __restrict__ xw, const float* __restrict__ dinv,
                           const int* __restrict__ row, const int* __restrict__ col,
                           float* __restrict__ h, int e) {
    int t = blockIdx.x * blockDim.x + threadIdx.x;
    int edge = t >> 5, lane = t & 31;
    if (edge >= e) return;
    int r = row[edge], c = col[edge];
    float nrm = dinv[r] * dinv[c];
    float4 v = ((const float4*)(xw + (size_t)r * HID))[lane];
    float* dst = h + (size_t)c * HID + lane * 4;
    atomicAdd(dst + 0, v.x * nrm);
    atomicAdd(dst + 1, v.y * nrm);
    atomicAdd(dst + 2, v.z * nrm);
    atomicAdd(dst + 3, v.w * nrm);
}

// h_f16 = relu(h + b)  (layer-1 output feeds only GEMM-2, so emit f16 directly)
__global__ void k_bias_relu_f16(const float* __restrict__ h, const float* __restrict__ b,
                                _Float16* __restrict__ outp, int n) {
    int idx = blockIdx.x * blockDim.x + threadIdx.x;
    if (idx >= n * 32) return;
    float4 v  = ((const float4*)h)[idx];
    float4 bb = ((const float4*)b)[idx & 31];
    v4h o = { (_Float16)fmaxf(v.x + bb.x, 0.0f),
              (_Float16)fmaxf(v.y + bb.y, 0.0f),
              (_Float16)fmaxf(v.z + bb.z, 0.0f),
              (_Float16)fmaxf(v.w + bb.w, 0.0f) };
    ((v4h*)outp)[idx] = o;
}

// h = relu(h + b) in place, f32 (layer-2 output feeds the pooling)
__global__ void k_bias_relu_f32(float* __restrict__ h, const float* __restrict__ b, int n) {
    int idx = blockIdx.x * blockDim.x + threadIdx.x;
    if (idx >= n * 32) return;
    float4 v  = ((float4*)h)[idx];
    float4 bb = ((const float4*)b)[idx & 31];
    v.x = fmaxf(v.x + bb.x, 0.0f);
    v.y = fmaxf(v.y + bb.y, 0.0f);
    v.z = fmaxf(v.z + bb.z, 0.0f);
    v.w = fmaxf(v.w + bb.w, 0.0f);
    ((float4*)h)[idx] = v;
}

// ---------------- mean pool + FC + log_softmax ----------------

__global__ void k_pool_init(float* __restrict__ pooled, float* __restrict__ cnt) {
    int i = blockIdx.x * blockDim.x + threadIdx.x;
    if (i < NGRF * HID) pooled[i] = 0.0f;
    if (i < NGRF) cnt[i] = 0.0f;
}

__global__ void k_pool_acc(const float* __restrict__ h, const int* __restrict__ batch,
                           float* __restrict__ pooled, float* __restrict__ cnt, int n) {
    int t = blockIdx.x * blockDim.x + threadIdx.x;
    int node = t >> 5, lane = t & 31;
    if (node >= n) return;
    int g = batch[node];
    float4 v = ((const float4*)h)[t];
    float* dst = pooled + (size_t)g * HID + lane * 4;
    atomicAdd(dst + 0, v.x);
    atomicAdd(dst + 1, v.y);
    atomicAdd(dst + 2, v.z);
    atomicAdd(dst + 3, v.w);
    if (lane == 0) atomicAdd(&cnt[g], 1.0f);
}

__global__ void k_fc_logsoftmax(const float* __restrict__ pooled, const float* __restrict__ cnt,
                                const float* __restrict__ Wfc, const float* __restrict__ bfc,
                                float* __restrict__ out) {
    int g = blockIdx.x * blockDim.x + threadIdx.x;
    if (g >= NGRF) return;
    float inv = 1.0f / fmaxf(cnt[g], 1.0f);
    float l[OUTC];
#pragma unroll
    for (int o = 0; o < OUTC; ++o) l[o] = bfc[o];
    for (int k = 0; k < HID; ++k) {
        float p = pooled[(size_t)g * HID + k] * inv;
#pragma unroll
        for (int o = 0; o < OUTC; ++o) l[o] += p * Wfc[k * OUTC + o];
    }
    float m = l[0];
#pragma unroll
    for (int o = 1; o < OUTC; ++o) m = fmaxf(m, l[o]);
    float s = 0.0f;
#pragma unroll
    for (int o = 0; o < OUTC; ++o) s += expf(l[o] - m);
    float lse = logf(s) + m;
#pragma unroll
    for (int o = 0; o < OUTC; ++o) out[(size_t)g * OUTC + o] = l[o] - lse;
}

// ---------------- launch ----------------

static inline int cdiv(long long a, int b) { return (int)((a + b - 1) / b); }

extern "C" void kernel_launch(void* const* d_in, const int* in_sizes, int n_in,
                              void* d_out, int out_size, void* d_ws, size_t ws_size,
                              hipStream_t stream) {
    const float* x   = (const float*)d_in[0];
    const int*   ei  = (const int*)d_in[1];
    const int*   bat = (const int*)d_in[2];
    const float* W1  = (const float*)d_in[3];
    const float* b1  = (const float*)d_in[4];
    const float* W2  = (const float*)d_in[5];
    const float* b2  = (const float*)d_in[6];
    const float* Wfc = (const float*)d_in[7];
    const float* bfc = (const float*)d_in[8];
    float* out = (float*)d_out;

    const int N = in_sizes[0] / HID;
    const int E = in_sizes[1] / 2;
    const int* row = ei;
    const int* col = ei + E;

    // workspace (float units): dinv[N] | bufA[N*128] | bufB[N*128] |
    //                          xh[N*128 f16 = N*64 f32] | Wt[128*128 f16] | pooled | cnt
    float* ws     = (float*)d_ws;
    float* dinv   = ws;
    float* bufA   = ws + (((size_t)N + 255) & ~(size_t)255);
    float* bufB   = bufA + (size_t)N * HID;
    _Float16* xh  = (_Float16*)(bufB + (size_t)N * HID);
    _Float16* Wt  = xh + (size_t)N * HID;
    float* pooled = (float*)(Wt + (size_t)HID * HID);
    float* cnt    = pooled + (size_t)NGRF * HID;

    const int T = 256;
    const long long NH4 = (long long)N * 32;   // N*128/4

    // normalization: dinv = rsqrt(1 + indegree)
    k_deg_init<<<cdiv(N, T), T, 0, stream>>>(dinv, N);
    k_deg_count<<<cdiv(E, T), T, 0, stream>>>(dinv, col, E);
    k_rsqrt_inplace<<<cdiv(N, T), T, 0, stream>>>(dinv, N);

    dim3 ggrid(cdiv(N, 64), HID / 32);

    // layer 1: xw1 = x@W1 (WMMA) ; h1 = Dsym(A+I) xw1 ; xh = relu_f16(h1 + b1)
    k_f32_to_f16<<<cdiv(NH4, T), T, 0, stream>>>(x, xh, (int)NH4);
    k_w_transpose_f16<<<cdiv(HID * HID, T), T, 0, stream>>>(W1, Wt);
    k_gemm_wmma<<<ggrid, T, 0, stream>>>(xh, Wt, bufA, N);
    k_self_init<<<cdiv(NH4, T), T, 0, stream>>>(bufA, dinv, bufB, N);
    k_edge_agg<<<cdiv((long long)E * 32, T), T, 0, stream>>>(bufA, dinv, row, col, bufB, E);
    k_bias_relu_f16<<<cdiv(NH4, T), T, 0, stream>>>(bufB, b1, xh, N);

    // layer 2: xw2 = h1@W2 (WMMA) ; h2 = Dsym(A+I) xw2 ; relu_f32(h2 + b2)
    k_w_transpose_f16<<<cdiv(HID * HID, T), T, 0, stream>>>(W2, Wt);
    k_gemm_wmma<<<ggrid, T, 0, stream>>>(xh, Wt, bufA, N);
    k_self_init<<<cdiv(NH4, T), T, 0, stream>>>(bufA, dinv, bufB, N);
    k_edge_agg<<<cdiv((long long)E * 32, T), T, 0, stream>>>(bufA, dinv, row, col, bufB, E);
    k_bias_relu_f32<<<cdiv(NH4, T), T, 0, stream>>>(bufB, b2, N);

    // mean pool + FC + log_softmax
    k_pool_init<<<cdiv(NGRF * HID, T), T, 0, stream>>>(pooled, cnt);
    k_pool_acc<<<cdiv(NH4, T), T, 0, stream>>>(bufB, bat, pooled, cnt, N);
    k_fc_logsoftmax<<<1, T, 0, stream>>>(pooled, cnt, Wfc, bfc, out);
}
